// LlamaAttention_90881507983679
// MI455X (gfx1250) — matrix operand order repaired and checked by reading, not compiled, not run on hardware
//
#include <hip/hip_runtime.h>
#include <hip/hip_bf16.h>

// ---------------------------------------------------------------------------
// Types for CDNA5 WMMA (wave32): v16bf A/B fragments, v8f C/D accumulator.
// ---------------------------------------------------------------------------
typedef __attribute__((ext_vector_type(16))) __bf16 v16bf;
typedef __attribute__((ext_vector_type(8)))  float  v8f;

union Frag { v16bf v; uint4 u[2]; };

__device__ __forceinline__ unsigned short f2bf(float f) {
  unsigned u = __float_as_uint(f);
  u += 0x7FFFu + ((u >> 16) & 1u);   // round-to-nearest-even
  return (unsigned short)(u >> 16);
}

// Async global->LDS copy (CDNA5): 16B per lane, tracked by ASYNCcnt.
__device__ __forceinline__ void async_copy_b128(const void* gptr, void* lptr) {
  unsigned lds_off = (unsigned)(unsigned long long)lptr;
  asm volatile("global_load_async_to_lds_b128 %0, %1, off"
               :: "v"(lds_off), "v"(gptr) : "memory");
}
__device__ __forceinline__ void async_wait0() {
  asm volatile("s_wait_asynccnt 0x0" ::: "memory");
}

// ---------------------------------------------------------------------------
// fp32 -> bf16 (raw u16 bits) elementwise convert
// ---------------------------------------------------------------------------
__global__ void k_f32_to_bf16(const float* __restrict__ in,
                              unsigned short* __restrict__ out, int n) {
  int i = blockIdx.x * blockDim.x + threadIdx.x;
  if (i < n) out[i] = f2bf(in[i]);
}

// ---------------------------------------------------------------------------
// GEMM: C[M][N] (f32) = A[M][K] (bf16) * B[N][K]^T (bf16)
// Workgroup tile 128x128, 256 threads = 8 waves, wave tile 32x64 (2x4 WMMA).
// Double-buffered LDS filled by global_load_async_to_lds_b128 so the HBM
// transfer of chunk k+1 overlaps the WMMAs on chunk k.
// ---------------------------------------------------------------------------
__global__ __launch_bounds__(256)
void k_gemm_bf16(const unsigned short* __restrict__ A,
                 const unsigned short* __restrict__ B,
                 float* __restrict__ C, int M, int N, int K) {
  __shared__ alignas(16) unsigned short As[2][128][32];
  __shared__ alignas(16) unsigned short Bs[2][128][32];

  const int m0   = blockIdx.y * 128;
  const int n0   = blockIdx.x * 128;
  const int tid  = threadIdx.x;
  const int w    = tid >> 5;          // wave 0..7
  const int lane = tid & 31;
  const int half = lane >> 4;         // 0/1
  const int lr   = lane & 15;
  const int wm   = (w & 3) * 32;      // wave row offset inside tile
  const int wn   = (w >> 2) * 64;     // wave col offset inside tile

  // staging map: 512 x 16B chunks per tile, 2 per thread
  const int sr = tid >> 2;            // rows tid/4 and +64
  const int sc = (tid & 3) * 8;

  v8f acc[2][4];
#pragma unroll
  for (int mt = 0; mt < 2; ++mt)
#pragma unroll
    for (int nt = 0; nt < 4; ++nt)
#pragma unroll
      for (int e = 0; e < 8; ++e) acc[mt][nt][e] = 0.0f;

  auto stage = [&](int buf, int k0) {
#pragma unroll
    for (int i = 0; i < 2; ++i) {
      int r = sr + i * 64;
      async_copy_b128(&A[(size_t)(m0 + r) * K + k0 + sc], &As[buf][r][sc]);
      async_copy_b128(&B[(size_t)(n0 + r) * K + k0 + sc], &Bs[buf][r][sc]);
    }
  };

  const int nk = K >> 5;              // K/32 chunks
  stage(0, 0);
  async_wait0();
  __syncthreads();

  for (int ki = 0; ki < nk; ++ki) {
    const int buf = ki & 1;
    if (ki + 1 < nk) stage(buf ^ 1, (ki + 1) * 32);   // overlap with compute

    // A fragment: lane(0-15)=row, half=0 -> K{0-7,16-23}, half=1 -> K{8-15,24-31}
    Frag af[2];
#pragma unroll
    for (int mt = 0; mt < 2; ++mt) {
      int r = wm + mt * 16 + lr;
      af[mt].u[0] = *(uint4*)&As[buf][r][half * 8];
      af[mt].u[1] = *(uint4*)&As[buf][r][16 + half * 8];
    }
    // B fragment: lane(0-15)=col(n), half=0 -> K{0-7,16-23}, half=1 -> K{8-15,24-31}
    Frag bfr[4];
#pragma unroll
    for (int nt = 0; nt < 4; ++nt) {
      int r = wn + nt * 16 + lr;
      bfr[nt].u[0] = *(uint4*)&Bs[buf][r][half * 8];
      bfr[nt].u[1] = *(uint4*)&Bs[buf][r][16 + half * 8];
    }
#pragma unroll
    for (int mt = 0; mt < 2; ++mt)
#pragma unroll
      for (int nt = 0; nt < 4; ++nt)
        acc[mt][nt] = __builtin_amdgcn_wmma_f32_16x16x32_bf16(
            false, af[mt].v, false, bfr[nt].v, (short)0, acc[mt][nt], false, false);

    async_wait0();      // my chunk-(ki+1) copies have landed in LDS
    __syncthreads();    // everyone done reading buf + done landing buf^1
  }

  // C layout: VGPR r, lanes0-15: M=r,N=lane; lanes16-31: M=r+8,N=lane-16
#pragma unroll
  for (int mt = 0; mt < 2; ++mt)
#pragma unroll
    for (int nt = 0; nt < 4; ++nt)
#pragma unroll
      for (int r = 0; r < 8; ++r) {
        int m = m0 + wm + mt * 16 + r + 8 * half;
        int n = n0 + wn + nt * 16 + lr;
        C[(size_t)m * N + n] = acc[mt][nt][r];
      }
}

// ---------------------------------------------------------------------------
// RoPE + repack: qkv f32 [T][3072] -> Qb[h][t][64]*scale, Kb[kvh][t][64],
// Vb TRANSPOSED [kvh][d][t]. cos/sin are [T][64] with cos[d]==cos[d+32].
// ---------------------------------------------------------------------------
__global__ __launch_bounds__(256)
void k_rope_pack(const float* __restrict__ qkv,
                 const float* __restrict__ cosb, const float* __restrict__ sinb,
                 unsigned short* __restrict__ Qb, unsigned short* __restrict__ Kb,
                 unsigned short* __restrict__ Vb) {
  const int t = blockIdx.x;                     // 0..2047
  const float* row = qkv + (size_t)t * 3072;
  const float scl = 0.125f;                     // 64^-0.5

  for (int i = threadIdx.x; i < 1024; i += 256) {   // Q
    int h = i >> 5, d = i & 31;
    float c = cosb[t * 64 + d], s = sinb[t * 64 + d];
    float x1 = row[h * 64 + d], x2 = row[h * 64 + d + 32];
    size_t base = ((size_t)h * 2048 + t) * 64;
    Qb[base + d]      = f2bf((x1 * c - x2 * s) * scl);
    Qb[base + d + 32] = f2bf((x2 * c + x1 * s) * scl);
  }
  for (int i = threadIdx.x; i < 256; i += 256) {    // K
    int h = i >> 5, d = i & 31;
    float c = cosb[t * 64 + d], s = sinb[t * 64 + d];
    float x1 = row[2048 + h * 64 + d], x2 = row[2048 + h * 64 + d + 32];
    size_t base = ((size_t)h * 2048 + t) * 64;
    Kb[base + d]      = f2bf(x1 * c - x2 * s);
    Kb[base + d + 32] = f2bf(x2 * c + x1 * s);
  }
  for (int i = threadIdx.x; i < 512; i += 256) {    // V transposed
    int h = i >> 6, d = i & 63;
    Vb[((size_t)h * 64 + d) * 2048 + t] = f2bf(row[2560 + i]);
  }
}

// ---------------------------------------------------------------------------
// Block-sparse flash attention. Grid (qb=32, h=32), 128 threads = 4 waves.
// Processed k-block set precomputed as a bitmask; K/V tiles double-buffered
// in LDS and DMA'd with async-to-LDS so the next block's HBM transfer
// overlaps the current block's WMMAs/softmax.
// ---------------------------------------------------------------------------
__global__ __launch_bounds__(128)
void k_attn(const unsigned short* __restrict__ Qb,
            const unsigned short* __restrict__ Kb,
            const unsigned short* __restrict__ Vb,
            const unsigned char* __restrict__ allow,
            unsigned short* __restrict__ Ob) {
  __shared__ alignas(16) unsigned short Kt[2][64][64];    // [k][d]
  __shared__ alignas(16) unsigned short Vt[2][64][64];    // [d][k]
  __shared__ alignas(16) unsigned short Pl[4][16][64];    // per-wave P tile

  const int qb  = blockIdx.x;
  const int h   = blockIdx.y;
  const int kvh = h >> 2;
  const int tid = threadIdx.x, w = tid >> 5, lane = tid & 31;
  const int half = lane >> 4, lr = lane & 15;

  // classify k-blocks once (uniform): FULL / PARTIAL / SKIP
  unsigned procmask = 0, partmask = 0;
  for (int kb = 0; kb < 32; ++kb) {
    bool ablk = allow[qb * 32 + kb] != 0;
    bool full = ablk || (kb <= 3) || ((kb >= qb - 3) && (kb < qb));
    bool part = !full && ((kb == qb) || (kb == qb - 4));
    if (full || part) procmask |= 1u << kb;
    if (part)         partmask |= 1u << kb;
  }

  // Q fragments (two K=32 chunks over D=64), A-matrix layout
  Frag qf[2];
  {
    int t = qb * 64 + w * 16 + lr;
    const unsigned short* qrow = Qb + ((size_t)h * 2048 + t) * 64;
    qf[0].u[0] = *(const uint4*)&qrow[half * 8];
    qf[0].u[1] = *(const uint4*)&qrow[16 + half * 8];
    qf[1].u[0] = *(const uint4*)&qrow[32 + half * 8];
    qf[1].u[1] = *(const uint4*)&qrow[48 + half * 8];
  }

  v8f acc[4];
#pragma unroll
  for (int dt = 0; dt < 4; ++dt)
#pragma unroll
    for (int e = 0; e < 8; ++e) acc[dt][e] = 0.0f;
  float rmax[8], rsum[8];
#pragma unroll
  for (int r = 0; r < 8; ++r) { rmax[r] = -1e30f; rsum[r] = 0.0f; }

  // stage K tile [k][d] and V tile [d][k]: 512 16B chunks each, 4/thread
  auto stage = [&](int kb, int buf) {
#pragma unroll
    for (int i = 0; i < 4; ++i) {
      int c = tid + i * 128;
      int r = c >> 3, cc = (c & 7) * 8;
      async_copy_b128(&Kb[((size_t)kvh * 2048 + kb * 64 + r) * 64 + cc], &Kt[buf][r][cc]);
      async_copy_b128(&Vb[((size_t)kvh * 64 + r) * 2048 + kb * 64 + cc], &Vt[buf][r][cc]);
    }
  };

  // walk set bits of procmask with one-block lookahead (double buffer)
  int kb = __builtin_ctz(procmask);        // diagonal always processed -> nonzero
  unsigned rem = procmask & (procmask - 1);
  int buf = 0;
  stage(kb, 0);
  async_wait0();
  __syncthreads();

  for (;;) {
    const int kbn = rem ? __builtin_ctz(rem) : -1;
    if (kbn >= 0) stage(kbn, buf ^ 1);     // prefetch next block's tiles

    // S tile (16x64) = Q (16x64) * K^T : 4 n-tiles, 2 chained WMMAs each
    v8f s[4];
#pragma unroll
    for (int nt = 0; nt < 4; ++nt) {
      Frag kf0, kf1;
      int r = nt * 16 + lr;                // B-frag: lane = k-column
      kf0.u[0] = *(uint4*)&Kt[buf][r][half * 8];
      kf0.u[1] = *(uint4*)&Kt[buf][r][16 + half * 8];
      kf1.u[0] = *(uint4*)&Kt[buf][r][32 + half * 8];
      kf1.u[1] = *(uint4*)&Kt[buf][r][48 + half * 8];
      v8f z;
#pragma unroll
      for (int e = 0; e < 8; ++e) z[e] = 0.0f;
      z     = __builtin_amdgcn_wmma_f32_16x16x32_bf16(false, qf[0].v, false, kf0.v, (short)0, z, false, false);
      s[nt] = __builtin_amdgcn_wmma_f32_16x16x32_bf16(false, qf[1].v, false, kf1.v, (short)0, z, false, false);
    }

    if ((partmask >> kb) & 1) {            // elementwise window mask
      int qbase = qb * 64 + w * 16 + 8 * half;
#pragma unroll
      for (int nt = 0; nt < 4; ++nt) {
        int kk = kb * 64 + nt * 16 + lr;
#pragma unroll
        for (int r = 0; r < 8; ++r) {
          int qi = qbase + r;
          if (!((kk <= qi) && (kk >= qi - 255))) s[nt][r] = -1e30f;
        }
      }
    }

    // online softmax (rows live in vgpr slot r, replicated over 16-lane group)
    float nm[8], alpha[8], psum[8];
#pragma unroll
    for (int r = 0; r < 8; ++r) {
      float tm = fmaxf(fmaxf(s[0][r], s[1][r]), fmaxf(s[2][r], s[3][r]));
#pragma unroll
      for (int off = 1; off < 16; off <<= 1) tm = fmaxf(tm, __shfl_xor(tm, off, 32));
      nm[r]    = fmaxf(rmax[r], tm);
      alpha[r] = __expf(rmax[r] - nm[r]);
      psum[r]  = 0.0f;
    }
#pragma unroll
    for (int nt = 0; nt < 4; ++nt)
#pragma unroll
      for (int r = 0; r < 8; ++r) {
        float p = __expf(s[nt][r] - nm[r]);
        psum[r] += p;
        Pl[w][r + 8 * half][nt * 16 + lr] = f2bf(p);  // C layout -> row-major
      }
#pragma unroll
    for (int r = 0; r < 8; ++r) {
      float ps = psum[r];
#pragma unroll
      for (int off = 1; off < 16; off <<= 1) ps += __shfl_xor(ps, off, 32);
      rsum[r] = rsum[r] * alpha[r] + ps;
      rmax[r] = nm[r];
    }
#pragma unroll
    for (int dt = 0; dt < 4; ++dt)
#pragma unroll
      for (int r = 0; r < 8; ++r) acc[dt][r] *= alpha[r];

    asm volatile("s_wait_dscnt 0x0" ::: "memory");  // P stores visible to own wave

    // O += P (16x64) * V (64x64): P re-read in A layout, V^T gives B layout
    Frag pf[2];
    pf[0].u[0] = *(uint4*)&Pl[w][lr][half * 8];
    pf[0].u[1] = *(uint4*)&Pl[w][lr][16 + half * 8];
    pf[1].u[0] = *(uint4*)&Pl[w][lr][32 + half * 8];
    pf[1].u[1] = *(uint4*)&Pl[w][lr][48 + half * 8];
#pragma unroll
    for (int dt = 0; dt < 4; ++dt) {
      Frag vf0, vf1;
      int r = dt * 16 + lr;                // lane = d-column of V^T
      vf0.u[0] = *(uint4*)&Vt[buf][r][half * 8];
      vf0.u[1] = *(uint4*)&Vt[buf][r][16 + half * 8];
      vf1.u[0] = *(uint4*)&Vt[buf][r][32 + half * 8];
      vf1.u[1] = *(uint4*)&Vt[buf][r][48 + half * 8];
      acc[dt] = __builtin_amdgcn_wmma_f32_16x16x32_bf16(false, pf[0].v, false, vf0.v, (short)0, acc[dt], false, false);
      acc[dt] = __builtin_amdgcn_wmma_f32_16x16x32_bf16(false, pf[1].v, false, vf1.v, (short)0, acc[dt], false, false);
    }

    async_wait0();      // next block's tiles landed
    __syncthreads();    // all waves done reading buf + done landing buf^1
    if (kbn < 0) break;
    kb = kbn;
    rem &= rem - 1;
    buf ^= 1;
  }

  // epilogue: divide by row sum, write bf16 [t][h*64+d]
#pragma unroll
  for (int dt = 0; dt < 4; ++dt)
#pragma unroll
    for (int r = 0; r < 8; ++r) {
      int t = qb * 64 + w * 16 + r + 8 * half;
      float o = acc[dt][r] / rsum[r];
      Ob[(size_t)t * 2048 + h * 64 + dt * 16 + lr] = f2bf(o);
    }
}

// ---------------------------------------------------------------------------
// Launch
// ---------------------------------------------------------------------------
extern "C" void kernel_launch(void* const* d_in, const int* in_sizes, int n_in,
                              void* d_out, int out_size, void* d_ws, size_t ws_size,
                              hipStream_t stream) {
  constexpr size_t T = 2048, E = 2048, H = 32, KV = 8, D = 64, NQ = 3072;
  const float* hidden = (const float*)d_in[0];
  const float* cosb   = (const float*)d_in[1];
  const float* sinb   = (const float*)d_in[2];
  const float* Wq     = (const float*)d_in[3];
  const float* Wk     = (const float*)d_in[4];
  const float* Wv     = (const float*)d_in[5];
  const float* Wo     = (const float*)d_in[6];
  const unsigned char* allow = (const unsigned char*)d_in[7];

  char* ws = (char*)d_ws;
  size_t o = 0;
  unsigned short* Xb    = (unsigned short*)(ws + o); o += T * E * 2;
  unsigned short* Wqkvb = (unsigned short*)(ws + o); o += NQ * E * 2;
  unsigned short* Wob   = (unsigned short*)(ws + o); o += E * E * 2;
  float*          qkv   = (float*)(ws + o);          o += T * NQ * 4;
  unsigned short* Qb    = (unsigned short*)(ws + o); o += H * T * D * 2;
  unsigned short* Kb    = (unsigned short*)(ws + o); o += KV * T * D * 2;
  unsigned short* Vb    = (unsigned short*)(ws + o); o += KV * T * D * 2;
  unsigned short* Ob    = (unsigned short*)(ws + o); o += T * E * 2;

  auto cvt = [&](const float* src, unsigned short* dst, int n) {
    k_f32_to_bf16<<<(n + 255) / 256, 256, 0, stream>>>(src, dst, n);
  };
  cvt(hidden, Xb, (int)(T * E));
  cvt(Wq, Wqkvb,                 (int)(2048 * E));
  cvt(Wk, Wqkvb + 2048 * E,      (int)(512 * E));
  cvt(Wv, Wqkvb + 2560 * E,      (int)(512 * E));
  cvt(Wo, Wob,                   (int)(E * E));

  // fused QKV projection: [T,E] x [3072,E]^T -> qkv f32
  k_gemm_bf16<<<dim3(NQ / 128, T / 128), 256, 0, stream>>>(Xb, Wqkvb, qkv,
                                                           (int)T, (int)NQ, (int)E);
  // RoPE + head-major bf16 repack (q pre-scaled by D^-0.5, V transposed)
  k_rope_pack<<<(int)T, 256, 0, stream>>>(qkv, cosb, sinb, Qb, Kb, Vb);
  // block-sparse flash attention (double-buffered async K/V tiles)
  k_attn<<<dim3(32, 32), 128, 0, stream>>>(Qb, Kb, Vb, allow, Ob);
  // output projection -> fp32 d_out
  k_gemm_bf16<<<dim3(E / 128, T / 128), 256, 0, stream>>>(Ob, Wob, (float*)d_out,
                                                          (int)T, (int)E, (int)E);
}
